// TrSFRNNAgent_32341103739524
// MI455X (gfx1250) — matrix operand order repaired and checked by reading, not compiled, not run on hardware
//
#include <hip/hip_runtime.h>
#include <hip/hip_bf16.h>

// ---------------------------------------------------------------------------
// Problem constants
// ---------------------------------------------------------------------------
#define BS    1024
#define N_AG  8
#define BSN   (BS * N_AG)      // 8192
#define D_IN  150
#define N_EN  8
#define N_AL  7
#define N_NA  6
#define N_ACT 14
#define PHI_N 32
#define PH_D  64
#define TE_D  64

typedef __attribute__((ext_vector_type(16))) __bf16          v16bf;
typedef __attribute__((ext_vector_type(16))) float           v16f;
typedef __attribute__((ext_vector_type(8)))  unsigned int    v8u;
typedef __attribute__((ext_vector_type(8)))  float           v8f;

__device__ __forceinline__ unsigned int f2bf(float x) {
  unsigned int u = __builtin_bit_cast(unsigned int, x);
  unsigned int r = u + 0x7FFFu + ((u >> 16) & 1u);   // round-to-nearest-even
  return r >> 16;
}
__device__ __forceinline__ unsigned int pack2bf(float lo, float hi) {
  return f2bf(lo) | (f2bf(hi) << 16);
}

// ---------------------------------------------------------------------------
// bf16-input / f32-accumulate WMMA GEMM:  C[M,N] = act(A[M,K] @ W[K,N] + bias)
// Block = 256 threads = 8 waves; block tile 128(M) x 64(N).
// Each wave owns 16 rows and all 64 cols -> 4 accumulators, 4 WMMAs / k-step.
// B panel (32 x 64) staged in LDS as packed bf16 pairs in contiguous-k (k = 16*half + s)
// B-fragment order, shared by all 8 waves; each lane reads its fragment with
// two 16B LDS vector loads (ds_load_b128).
// Launcher guarantees M % 128 == 0 and N % 64 == 0. Ragged K handled by a
// single checked tail step; the main loop is branch-free.
// act: 0 = none, 1 = leaky_relu(0.01)
// ---------------------------------------------------------------------------
__global__ void __launch_bounds__(256)
gemm_bf16_wmma(const float* __restrict__ A, int lda,
               const float* __restrict__ W, int ldw,
               const float* __restrict__ bias,
               float* __restrict__ C, int ldc,
               int M, int N, int K, int act)
{
  const int wave = threadIdx.x >> 5;
  const int lane = threadIdx.x & 31;
  const int half = lane >> 4;               // 0: lanes 0-15, 1: lanes 16-31
  const int l16  = lane & 15;
  const int m0     = blockIdx.y * 128 + wave * 16;
  const int n_base = blockIdx.x * 64;
  const int row    = m0 + l16;              // A row handled by this lane

  // LDS B panel: u32[n][kp], n = 0..63, kp = 0..15 (bf16 pair k=2kp, 2kp+1)
  __shared__ unsigned int sB[64 * 16];

  v8f acc0 = {}, acc1 = {}, acc2 = {}, acc3 = {};
  const int kfull = K & ~31;

  // ---------------- main loop: no bounds checks ----------------
  for (int kb = 0; kb < kfull; kb += 32) {
    // stage B: 1024 u32, 4 per thread; rows of W are coalesced across threads.
    // Prefetch the next panel unconditionally: prefetch is speculative per the
    // CDNA5 ISA (invalid translations silently dropped), so overshoot is safe.
    #pragma unroll
    for (int it = 0; it < 4; ++it) {
      int idx = threadIdx.x + (it << 8);            // 0..1023
      int kp = idx >> 6, n = idx & 63;
      const float* wp = W + (long)(kb + 2 * kp) * ldw + n_base + n;
      sB[n * 16 + kp] = pack2bf(wp[0], wp[ldw]);
      __builtin_prefetch(wp + 32 * (long)ldw, 0, 1);
    }
    __syncthreads();

    // A fragment: element i -> k = kb + half*16 + i
    v16f afv;
    if ((lda & 3) == 0) {
      const float4* p0 = (const float4*)(A + (long)row * lda + kb + (half << 4));
      const float4* p1 = (const float4*)(A + (long)row * lda + kb + 8 + (half << 4));
      float4 a0 = p0[0], a1 = p0[1], b0 = p1[0], b1 = p1[1];
      afv[0]=a0.x; afv[1]=a0.y; afv[2]=a0.z; afv[3]=a0.w;
      afv[4]=a1.x; afv[5]=a1.y; afv[6]=a1.z; afv[7]=a1.w;
      afv[8]=b0.x; afv[9]=b0.y; afv[10]=b0.z; afv[11]=b0.w;
      afv[12]=b1.x; afv[13]=b1.y; afv[14]=b1.z; afv[15]=b1.w;
    } else {
      #pragma unroll
      for (int i = 0; i < 16; ++i)
        afv[i] = A[(long)row * lda + kb + (half << 4) + i];
    }
    v16bf a = __builtin_convertvector(afv, v16bf);   // packed f32->bf16 cvt

    // 4 column tiles: fragment = 8 consecutive u32 at [ (nt*16+l16)*16 + half*8 ]
    #pragma unroll
    for (int nt = 0; nt < 4; ++nt) {
      const uint4* bp = (const uint4*)&sB[((nt << 4) + l16) * 16 + (half << 3)];
      uint4 q0 = bp[0], q1 = bp[1];
      v8u bw; bw[0]=q0.x; bw[1]=q0.y; bw[2]=q0.z; bw[3]=q0.w;
              bw[4]=q1.x; bw[5]=q1.y; bw[6]=q1.z; bw[7]=q1.w;
      v16bf b = __builtin_bit_cast(v16bf, bw);
      if (nt == 0) acc0 = __builtin_amdgcn_wmma_f32_16x16x32_bf16(false, a, false, b, (short)0, acc0, false, false);
      if (nt == 1) acc1 = __builtin_amdgcn_wmma_f32_16x16x32_bf16(false, a, false, b, (short)0, acc1, false, false);
      if (nt == 2) acc2 = __builtin_amdgcn_wmma_f32_16x16x32_bf16(false, a, false, b, (short)0, acc2, false, false);
      if (nt == 3) acc3 = __builtin_amdgcn_wmma_f32_16x16x32_bf16(false, a, false, b, (short)0, acc3, false, false);
    }
    __syncthreads();
  }

  // ---------------- tail: ragged K with zero fill ----------------
  if (kfull < K) {
    const int kb = kfull;
    #pragma unroll
    for (int it = 0; it < 4; ++it) {
      int idx = threadIdx.x + (it << 8);
      int kp = idx >> 6, n = idx & 63;
      int k0 = kb + 2 * kp;
      float w0 = (k0     < K) ? W[(long)k0 * ldw + n_base + n]       : 0.f;
      float w1 = (k0 + 1 < K) ? W[(long)(k0 + 1) * ldw + n_base + n] : 0.f;
      sB[n * 16 + kp] = pack2bf(w0, w1);
    }
    __syncthreads();
    v16f afv;
    #pragma unroll
    for (int i = 0; i < 16; ++i) {
      int k = kb + (half << 4) + i;
      afv[i] = (k < K) ? A[(long)row * lda + k] : 0.f;
    }
    v16bf a = __builtin_convertvector(afv, v16bf);
    #pragma unroll
    for (int nt = 0; nt < 4; ++nt) {
      const uint4* bp = (const uint4*)&sB[((nt << 4) + l16) * 16 + (half << 3)];
      uint4 q0 = bp[0], q1 = bp[1];
      v8u bw; bw[0]=q0.x; bw[1]=q0.y; bw[2]=q0.z; bw[3]=q0.w;
              bw[4]=q1.x; bw[5]=q1.y; bw[6]=q1.z; bw[7]=q1.w;
      v16bf b = __builtin_bit_cast(v16bf, bw);
      if (nt == 0) acc0 = __builtin_amdgcn_wmma_f32_16x16x32_bf16(false, a, false, b, (short)0, acc0, false, false);
      if (nt == 1) acc1 = __builtin_amdgcn_wmma_f32_16x16x32_bf16(false, a, false, b, (short)0, acc1, false, false);
      if (nt == 2) acc2 = __builtin_amdgcn_wmma_f32_16x16x32_bf16(false, a, false, b, (short)0, acc2, false, false);
      if (nt == 3) acc3 = __builtin_amdgcn_wmma_f32_16x16x32_bf16(false, a, false, b, (short)0, acc3, false, false);
    }
    __syncthreads();
  }

  // ---------------- epilogue: bias + activation + store ----------------
  #pragma unroll
  for (int nt = 0; nt < 4; ++nt) {
    v8f accv = (nt == 0) ? acc0 : (nt == 1) ? acc1 : (nt == 2) ? acc2 : acc3;
    int col = n_base + (nt << 4) + l16;
    float bv = bias ? bias[col] : 0.f;
    #pragma unroll
    for (int r = 0; r < 8; ++r) {
      int rr = m0 + r + half * 8;   // C layout: vgpr r -> M = r (+8 upper half)
      float v = accv[r] + bv;
      if (act == 1) v = (v > 0.f) ? v : 0.01f * v;
      C[(long)rr * ldc + col] = v;
    }
  }
}

// ---------------------------------------------------------------------------
// K0: build own_obs (own|agent_id|compact_last, padded to 32) and enemy_feats
// ---------------------------------------------------------------------------
__global__ void prep_kernel(const float* __restrict__ inputs,
                            const float* __restrict__ agent_id,
                            float* __restrict__ ownobs,
                            float* __restrict__ enf)
{
  int r = blockIdx.x, t = threadIdx.x;   // 64 threads
  const float* in = inputs + (long)r * D_IN;
  if (t < 16)       ownobs[r * 32 + t] = in[t];
  else if (t < 24)  ownobs[r * 32 + t] = agent_id[r * 8 + (t - 16)];
  else if (t < 30)  ownobs[r * 32 + t] = in[136 + (t - 24)];      // last_a[0..5]
  else if (t == 30) { float s = 0.f; for (int j = 0; j < 8; ++j) s += in[142 + j];
                      ownobs[r * 32 + 30] = s; }                   // sum(atk_info)
  else if (t == 31) ownobs[r * 32 + 31] = 0.f;
  { int e = t >> 3, f = t & 7;
    enf[r * 72 + e * 9 + f] = in[16 + e * 8 + f]; }               // enemy feats
  if (t < 8) enf[r * 72 + t * 9 + 8] = in[142 + t];               // attack column
}

// ---------------------------------------------------------------------------
// K0b: task embedding vector (64)
// ---------------------------------------------------------------------------
__global__ void taskemb_kernel(const float* __restrict__ tw,
                               const float* __restrict__ te_W,
                               const float* __restrict__ te_b,
                               float* __restrict__ out)
{
  int t = threadIdx.x;  // 64
  float v = te_b[t];
  for (int k = 0; k < PHI_N; ++k) v += tw[k] * te_W[k * TE_D + t];
  out[t] = v;
}

// ---------------------------------------------------------------------------
// K1: per-row own/q projections + ally & enemy attention -> attn_feature(192)
// ---------------------------------------------------------------------------
__global__ void attn_kernel(const float* __restrict__ inputs,
                            const float* __restrict__ ownobs,
                            const float* __restrict__ enf,
                            const float* own_W, const float* own_b,
                            const float* q_W,  const float* q_b,
                            const float* ak_W, const float* ak_b,
                            const float* av_W, const float* av_b,
                            const float* ek_W, const float* ek_b,
                            const float* ev_W, const float* ev_b,
                            float* __restrict__ attnf)
{
  int r = blockIdx.x, t = threadIdx.x;   // 64 threads
  __shared__ float sOwn[32], sQ[64], sV[8][64], sE[8];
  if (t < 31) sOwn[t] = ownobs[r * 32 + t];
  __syncthreads();
  float qv = q_b[t], of = own_b[t];
  for (int k = 0; k < 31; ++k) {
    float x = sOwn[k];
    qv += x * q_W[k * 64 + t];
    of += x * own_W[k * 64 + t];
  }
  sQ[t] = qv;
  attnf[(long)r * 192 + t] = of;
  __syncthreads();

  // ----- allies (7, feat 8) -----
  const float* al = inputs + (long)r * D_IN + 16 + 64;
  for (int n = 0; n < 7; ++n) {
    float v = av_b[t];
    for (int f = 0; f < 8; ++f) v += al[n * 8 + f] * av_W[f * 64 + t];
    sV[n][t] = v;
  }
  if (t < 7) {
    float e = 0.f;
    for (int a = 0; a < 64; ++a) {
      float kv = ak_b[a];
      for (int f = 0; f < 8; ++f) kv += al[t * 8 + f] * ak_W[f * 64 + a];
      e += sQ[a] * kv;
    }
    sE[t] = e * 0.125f;   // 1/sqrt(64)
  }
  __syncthreads();
  if (t == 0) {
    float mx = sE[0]; for (int n = 1; n < 7; ++n) mx = fmaxf(mx, sE[n]);
    float s = 0.f;
    for (int n = 0; n < 7; ++n) { float ex = __expf(sE[n] - mx); sE[n] = ex; s += ex; }
    float inv = 1.f / s;
    for (int n = 0; n < 7; ++n) sE[n] *= inv;
  }
  __syncthreads();
  { float af = 0.f; for (int n = 0; n < 7; ++n) af += sE[n] * sV[n][t];
    attnf[(long)r * 192 + 64 + t] = af; }
  __syncthreads();

  // ----- enemies (8, feat 9) -----
  const float* ef = enf + (long)r * 72;
  for (int n = 0; n < 8; ++n) {
    float v = ev_b[t];
    for (int f = 0; f < 9; ++f) v += ef[n * 9 + f] * ev_W[f * 64 + t];
    sV[n][t] = v;
  }
  if (t < 8) {
    float e = 0.f;
    for (int a = 0; a < 64; ++a) {
      float kv = ek_b[a];
      for (int f = 0; f < 9; ++f) kv += ef[t * 9 + f] * ek_W[f * 64 + a];
      e += sQ[a] * kv;
    }
    sE[t] = e * 0.125f;
  }
  __syncthreads();
  if (t == 0) {
    float mx = sE[0]; for (int n = 1; n < 8; ++n) mx = fmaxf(mx, sE[n]);
    float s = 0.f;
    for (int n = 0; n < 8; ++n) { float ex = __expf(sE[n] - mx); sE[n] = ex; s += ex; }
    float inv = 1.f / s;
    for (int n = 0; n < 8; ++n) sE[n] *= inv;
  }
  __syncthreads();
  { float eo = 0.f; for (int n = 0; n < 8; ++n) eo += sE[n] * sV[n][t];
    attnf[(long)r * 192 + 128 + t] = eo; }
}

// ---------------------------------------------------------------------------
// K4: GRU gates + assemble psi_in(320) / tau(264) + r_shaping reduction
// ---------------------------------------------------------------------------
__global__ void __launch_bounds__(256)
gru_kernel(const float* __restrict__ gi, const float* __restrict__ gh,
           const float* __restrict__ hprev, const float* __restrict__ cur_action,
           const float* __restrict__ taskemb,
           const float* __restrict__ rs_W, const float* __restrict__ rs_b,
           float* __restrict__ out_h, float* __restrict__ psiin,
           float* __restrict__ tau,   float* __restrict__ out_rs)
{
  int r = blockIdx.x, j = threadIdx.x;   // 256 threads
  __shared__ float sRed[256];
  long b0 = (long)r * 768;
  float ir = gi[b0 + j], iz = gi[b0 + 256 + j], in_ = gi[b0 + 512 + j];
  float hr = gh[b0 + j], hz = gh[b0 + 256 + j], hn  = gh[b0 + 512 + j];
  float rg = 1.f / (1.f + __expf(-(ir + hr)));
  float z  = 1.f / (1.f + __expf(-(iz + hz)));
  float n  = tanhf(in_ + rg * hn);
  float hp = hprev[(long)r * 256 + j];
  float h  = (1.f - z) * n + z * hp;
  out_h[(long)r * 256 + j] = h;
  psiin[(long)r * 320 + j] = h;
  tau[(long)r * 264 + j]   = h;
  if (j < 64) psiin[(long)r * 320 + 256 + j] = taskemb[j];
  if (j < 8) {
    const float* ca = cur_action + (long)r * N_ACT;
    if (j < 6)       tau[(long)r * 264 + 256 + j] = ca[j];
    else if (j == 6) { float s = 0.f; for (int a = 6; a < 14; ++a) s += ca[a];
                       tau[(long)r * 264 + 262] = s; }
    else             tau[(long)r * 264 + 263] = 0.f;
  }
  sRed[j] = h * rs_W[j];
  __syncthreads();
  for (int off = 128; off > 0; off >>= 1) {
    if (j < off) sRed[j] += sRed[j + off];
    __syncthreads();
  }
  if (j == 0) out_rs[r] = sRed[0] + rs_b[0];
}

// ---------------------------------------------------------------------------
// K9: psi assembly (wo_action + attack_psi), mean-centered advantage mask
// ---------------------------------------------------------------------------
__global__ void psi_kernel(const float* __restrict__ psi_hid,
                           const float* __restrict__ ee_out,
                           const float* woA_W, const float* woA_b,
                           const float* aa_W,  const float* aa_b,
                           const float* __restrict__ cur_action,
                           float* __restrict__ out_psi,
                           float* __restrict__ adv_mask)
{
  int r = blockIdx.x, t = threadIdx.x;   // 64 threads
  __shared__ float sEat[8], sH[64], sP[16], sCA[16], sHat;
  if (t < 8) {
    float e = 0.f;
    for (int d = 0; d < 64; ++d) e += ee_out[((long)r * 8 + t) * 64 + d] * aa_W[d];
    sEat[t] = e;
  }
  if (t < 14) sCA[t] = cur_action[(long)r * N_ACT + t];
  __syncthreads();
  float ab0 = aa_b[0];
  for (int p = 0; p < PHI_N; ++p) {
    sH[t] = psi_hid[(long)r * 2048 + p * 64 + t];
    __syncthreads();
    if (t < 6) {
      float w = woA_b[t];
      for (int d = 0; d < 64; ++d) w += sH[d] * woA_W[d * 6 + t];
      sP[t] = w;
    } else if (t == 6) {
      float hh = 0.f;
      for (int d = 0; d < 64; ++d) hh += sH[d] * aa_W[64 + d];
      sHat = hh;
    }
    __syncthreads();
    if (t < 14) {
      float val = (t < 6) ? sP[t] : (sEat[t - 6] + sHat + ab0);
      out_psi[(long)r * 448 + p * 14 + t] = val;
      sP[t] = val;
    }
    __syncthreads();
    if (t == 0) {
      float mean = 0.f; for (int a = 0; a < 14; ++a) mean += sP[a];
      mean *= (1.f / 14.f);
      float adv = 0.f; for (int a = 0; a < 14; ++a) adv += sCA[a] * (sP[a] - mean);
      adv_mask[(long)r * 32 + p] = (adv > 0.f) ? 1.f : 0.f;
    }
    __syncthreads();
  }
}

// ---------------------------------------------------------------------------
// K12: phi[r,p] = phi_hid[r,p,:] . pp_W + pp_b   (one wave per (r,p))
// ---------------------------------------------------------------------------
__global__ void __launch_bounds__(256)
phi_kernel(const float* __restrict__ phi_hid,
           const float* __restrict__ pp_W, const float* __restrict__ pp_b,
           float* __restrict__ out_phi)
{
  int idx  = blockIdx.x * 8 + (threadIdx.x >> 5);   // (r,p) flat, 8 waves/block
  int lane = threadIdx.x & 31;
  int r = idx >> 5, p = idx & 31;
  const float* src = phi_hid + (long)r * 2048 + p * 64;
  float s = src[lane] * pp_W[lane] + src[lane + 32] * pp_W[lane + 32];
  for (int off = 16; off > 0; off >>= 1) s += __shfl_down(s, off, 32);
  if (lane == 0) out_phi[idx] = s + pp_b[0];
}

// ---------------------------------------------------------------------------
// K13: group attention over 8 agents per (b, p) -> phi_tilde
// ---------------------------------------------------------------------------
__global__ void gattn_kernel(const float* __restrict__ phi_hid,
                             const float* __restrict__ adv_mask,
                             const float* Wq, const float* Wk, const float* Wv,
                             const float* pp_W, const float* pp_b,
                             float* __restrict__ out_phit)
{
  int p = blockIdx.x, b = blockIdx.y, t = threadIdx.x;   // 64 threads
  __shared__ float sPh[8][64], sQ[8][64], sK[8][64], sV[8][64], sEn[8][8], sM[8];
  for (int j = 0; j < 8; ++j)
    sPh[j][t] = phi_hid[((long)b * 8 + j) * 2048 + p * 64 + t];
  if (t < 8) sM[t] = adv_mask[((long)b * 8 + t) * 32 + p];
  __syncthreads();
  for (int j = 0; j < 8; ++j) {
    float aq = 0.f, ak = 0.f, av = 0.f;
    for (int k = 0; k < 64; ++k) {
      float x = sPh[j][k];
      aq += x * Wq[k * 64 + t];
      ak += x * Wk[k * 64 + t];
      av += x * Wv[k * 64 + t];
    }
    sQ[j][t] = aq; sK[j][t] = ak; sV[j][t] = av;
  }
  __syncthreads();
  { int i = t >> 3, j = t & 7;
    float e = 0.f;
    for (int k = 0; k < 64; ++k) e += sQ[i][k] * sK[j][k];
    e *= 0.125f;                                      // 1/sqrt(64)
    bool valid = (sM[j] > 0.f) || (i == j);
    sEn[i][j] = valid ? e : -1e9f; }
  __syncthreads();
  if (t < 8) {
    float mx = sEn[t][0]; for (int j = 1; j < 8; ++j) mx = fmaxf(mx, sEn[t][j]);
    float s = 0.f;
    for (int j = 0; j < 8; ++j) { float ex = __expf(sEn[t][j] - mx); sEn[t][j] = ex; s += ex; }
    float inv = 1.f / s;
    for (int j = 0; j < 8; ++j) sEn[t][j] *= inv;
  }
  __syncthreads();
  for (int i = 0; i < 8; ++i) {
    float pt = 0.f;
    for (int j = 0; j < 8; ++j) pt += sEn[i][j] * sV[j][t];
    sPh[i][t] = pt;    // reuse as phi_tilde_hid
  }
  __syncthreads();
  if (t < 8) {
    float v = pp_b[0];
    for (int k = 0; k < 64; ++k) v += sPh[t][k] * pp_W[k];
    out_phit[((long)b * 8 + t) * 32 + p] = v;
  }
}

// ---------------------------------------------------------------------------
// Host-side orchestration
// ---------------------------------------------------------------------------
extern "C" void kernel_launch(void* const* d_in, const int* in_sizes, int n_in,
                              void* d_out, int out_size, void* d_ws, size_t ws_size,
                              hipStream_t stream)
{
  const float* inputs     = (const float*)d_in[0];
  const float* cur_action = (const float*)d_in[1];
  const float* hidden     = (const float*)d_in[2];
  const float* task_w     = (const float*)d_in[3];
  const float* agent_id   = (const float*)d_in[4];
  const float* own_W = (const float*)d_in[5],  *own_b = (const float*)d_in[6];
  const float* q_W   = (const float*)d_in[7],  *q_b   = (const float*)d_in[8];
  const float* ak_W  = (const float*)d_in[9],  *ak_b  = (const float*)d_in[10];
  const float* ek_W  = (const float*)d_in[11], *ek_b  = (const float*)d_in[12];
  const float* av_W  = (const float*)d_in[13], *av_b  = (const float*)d_in[14];
  const float* ev_W  = (const float*)d_in[15], *ev_b  = (const float*)d_in[16];
  const float* gru_Wi = (const float*)d_in[17], *gru_Wh = (const float*)d_in[18];
  const float* gru_bi = (const float*)d_in[19], *gru_bh = (const float*)d_in[20];
  const float* te_W  = (const float*)d_in[21], *te_b  = (const float*)d_in[22];
  const float* psi_W1 = (const float*)d_in[23], *psi_b1 = (const float*)d_in[24];
  const float* psi_W2 = (const float*)d_in[25], *psi_b2 = (const float*)d_in[26];
  const float* woA_W = (const float*)d_in[27], *woA_b = (const float*)d_in[28];
  const float* ee_W1 = (const float*)d_in[29], *ee_b1 = (const float*)d_in[30];
  const float* ee_W2 = (const float*)d_in[31], *ee_b2 = (const float*)d_in[32];
  const float* aa_W  = (const float*)d_in[33], *aa_b  = (const float*)d_in[34];
  const float* pe_W1 = (const float*)d_in[35], *pe_b1 = (const float*)d_in[36];
  const float* pe_W2 = (const float*)d_in[37], *pe_b2 = (const float*)d_in[38];
  const float* pp_W  = (const float*)d_in[39], *pp_b  = (const float*)d_in[40];
  const float* ga_Wq = (const float*)d_in[41], *ga_Wk = (const float*)d_in[42];
  const float* ga_Wv = (const float*)d_in[43];
  const float* rs_W  = (const float*)d_in[44], *rs_b  = (const float*)d_in[45];

  // Workspace layout (floats)
  float* ws = (float*)d_ws;
  float* OWNOBS  = ws;                        // 8192*32
  float* ENF     = OWNOBS  + 262144;          // 8192*72
  float* ATTNF   = ENF     + 589824;          // 8192*192
  float* GI      = ATTNF   + 1572864;         // 8192*768
  float* GH      = GI      + 6291456;         // 8192*768
  float* PSIIN   = GH      + 6291456;         // 8192*320
  float* TAU     = PSIIN   + 2621440;         // 8192*264
  float* PSI_H1  = TAU     + 2162688;         // 8192*512
  float* PSI_HID = PSI_H1  + 4194304;         // 8192*2048
  float* EE_HID  = PSI_HID + 16777216;        // 65536*128
  float* EE_OUT  = EE_HID  + 8388608;         // 65536*64
  float* PE_H1   = EE_OUT  + 4194304;         // 8192*512
  float* PHI_HID = PE_H1   + 4194304;         // 8192*2048
  float* ADVMSK  = PHI_HID + 16777216;        // 8192*32
  float* TASKEMB = ADVMSK  + 262144;          // 64

  // Output layout (floats)
  float* out      = (float*)d_out;
  float* OUT_H    = out;                      // 8192*256
  float* OUT_PSI  = OUT_H    + 2097152;       // 8192*448
  float* OUT_PHI  = OUT_PSI  + 3670016;       // 8192*32
  float* OUT_PHIT = OUT_PHI  + 262144;        // 8192*32
  float* OUT_RS   = OUT_PHIT + 262144;        // 8192

  // All GEMMs guarantee M % 128 == 0 and N % 64 == 0.
  auto gemm = [&](const float* A, int lda, const float* W, int ldw,
                  const float* bias, float* C, int ldc,
                  int M, int N, int K, int act) {
    dim3 grid(N / 64, M / 128);
    gemm_bf16_wmma<<<grid, 256, 0, stream>>>(A, lda, W, ldw, bias, C, ldc, M, N, K, act);
  };

  // 1) prep + task embedding
  prep_kernel<<<BSN, 64, 0, stream>>>(inputs, agent_id, OWNOBS, ENF);
  taskemb_kernel<<<1, 64, 0, stream>>>(task_w, te_W, te_b, TASKEMB);

  // 2) per-row attention -> attn_feature
  attn_kernel<<<BSN, 64, 0, stream>>>(inputs, OWNOBS, ENF,
                                      own_W, own_b, q_W, q_b,
                                      ak_W, ak_b, av_W, av_b,
                                      ek_W, ek_b, ev_W, ev_b, ATTNF);

  // 3) GRU: gi / gh GEMMs then gate math (+ psi_in / tau assembly + r_shaping)
  gemm(ATTNF, 192, gru_Wi, 768, gru_bi, GI, 768, BSN, 768, 192, 0);
  gemm(hidden, 256, gru_Wh, 768, gru_bh, GH, 768, BSN, 768, 256, 0);
  gru_kernel<<<BSN, 256, 0, stream>>>(GI, GH, hidden, cur_action, TASKEMB,
                                      rs_W, rs_b, OUT_H, PSIIN, TAU, OUT_RS);

  // 4) psi branch
  gemm(PSIIN, 320, psi_W1, 512, psi_b1, PSI_H1, 512, BSN, 512, 320, 1);
  gemm(PSI_H1, 512, psi_W2, 2048, psi_b2, PSI_HID, 2048, BSN, 2048, 512, 0);
  gemm(ENF, 9, ee_W1, 128, ee_b1, EE_HID, 128, BSN * N_EN, 128, 9, 1);
  gemm(EE_HID, 128, ee_W2, 64, ee_b2, EE_OUT, 64, BSN * N_EN, 64, 128, 0);
  psi_kernel<<<BSN, 64, 0, stream>>>(PSI_HID, EE_OUT, woA_W, woA_b, aa_W, aa_b,
                                     cur_action, OUT_PSI, ADVMSK);

  // 5) pe branch: logvar half of pe_W2 is dead -> compute only first 2048 cols
  gemm(TAU, 264, pe_W1, 512, pe_b1, PE_H1, 512, BSN, 512, 263, 1);
  gemm(PE_H1, 512, pe_W2, 4096, pe_b2, PHI_HID, 2048, BSN, 2048, 512, 0);
  phi_kernel<<<(BSN * PHI_N) / 8, 256, 0, stream>>>(PHI_HID, pp_W, pp_b, OUT_PHI);

  // 6) group attention -> phi_tilde
  dim3 ggrid(PHI_N, BS);
  gattn_kernel<<<ggrid, 64, 0, stream>>>(PHI_HID, ADVMSK, ga_Wq, ga_Wk, ga_Wv,
                                         pp_W, pp_b, OUT_PHIT);
  (void)in_sizes; (void)n_in; (void)out_size; (void)ws_size;
}
